// GatedAttn_23510650978924
// MI455X (gfx1250) — matrix operand order repaired and checked by reading, not compiled, not run on hardware
//
#include <hip/hip_runtime.h>

// ---------------------------------------------------------------------------
// GatedAttn on gfx1250 (MI455X). wave32, v_wmma_f32_16x16x32_f16 everywhere.
// B=8, C=128, N=1024 (=32x32), heads=4, d=32, c_hidden=128.
// ---------------------------------------------------------------------------

#define Bb     8
#define Cc     128
#define Nn     1024
#define HEADS  4
#define Dd     32
#define CH     128      // c_hidden
#define JQKV   384      // 3 * c_hidden
#define JOUT   256      // 2 * C

typedef _Float16 h8  __attribute__((ext_vector_type(8)));
typedef _Float16 h16 __attribute__((ext_vector_type(16)));
typedef float    f8v __attribute__((ext_vector_type(8)));
typedef float    f4v __attribute__((ext_vector_type(4)));

static __device__ __forceinline__ f8v fzero8() {
    f8v z;
#pragma unroll
    for (int i = 0; i < 8; ++i) z[i] = 0.0f;
    return z;
}

static __device__ __forceinline__ h16 cat8(h8 lo, h8 hi) {
    return __builtin_shufflevector(lo, hi, 0,1,2,3,4,5,6,7,8,9,10,11,12,13,14,15);
}

// A fragment: 16x32 (MxK) f16, source row-major [row][k], stride ld halfs.
// Lane L (l16 = L&15, g = L>>4) element e maps to (M=l16, K=16*(e>>3)+8g+(e&7)):
// two contiguous 16B loads per lane.
static __device__ __forceinline__ h16 load_a_frag(const _Float16* __restrict__ base,
                                                  int ld, int row0, int k0, int lane) {
    const int l16 = lane & 15, g = lane >> 4;
    const _Float16* p = base + (size_t)(row0 + l16) * ld + k0 + 8 * g;
    h8 lo = *(const h8*)(p);
    h8 hi = *(const h8*)(p + 16);
    return cat8(lo, hi);
}

// B fragment: 32x16 (KxN) f16, sourced from B^T stored row-major [n][k], stride ld.
// Lane L element e maps to (N=l16, K=16*(e>>3)+8g+(e&7)): two contiguous 16B loads per lane.
static __device__ __forceinline__ h16 load_bT_frag(const _Float16* __restrict__ baseT,
                                                   int ld, int n0, int k0, int lane) {
    const int l16 = lane & 15, g = lane >> 4;
    const _Float16* p = baseT + (size_t)(n0 + l16) * ld + k0 + 8 * g;
    h8 lo = *(const h8*)(p);
    h8 hi = *(const h8*)(p + 16);
    return cat8(lo, hi);
}

static __device__ __forceinline__ f8v wmma_f16(h16 a, h16 b, f8v c) {
    return __builtin_amdgcn_wmma_f32_16x16x32_f16(false, a, false, b, (short)0, c,
                                                  false, false);
}

// ---------------------------------------------------------------------------
// Kernel 1: f32 -> f16 transposes.
//   xt[b][n][c]   = x[b][c][n]
//   qkvwT[j][c]   = qkv_w[c][j]   (j < 384)
//   gatewT[j][c]  = gate_w[c][j]  (j < 256)
// ---------------------------------------------------------------------------
#define PREP_X   (Bb * Nn * Cc)
#define PREP_QW  (JQKV * Cc)
#define PREP_GW  (JOUT * Cc)
#define PREP_TOT (PREP_X + PREP_QW + PREP_GW)

__global__ __launch_bounds__(256) void prep_kernel(const float* __restrict__ x,
                                                   const float* __restrict__ qkv_w,
                                                   const float* __restrict__ gate_w,
                                                   _Float16* __restrict__ xt,
                                                   _Float16* __restrict__ qkvwT,
                                                   _Float16* __restrict__ gatewT) {
    int i = blockIdx.x * 256 + threadIdx.x;
    if (i < PREP_X) {
        // i enumerates x contiguously: i = (b*Cc + c)*Nn + n  -> coalesced reads
        int n = i & (Nn - 1);
        int c = (i >> 10) & (Cc - 1);
        int b = i >> 17;
        xt[((size_t)b * Nn + n) * Cc + c] = (_Float16)x[i];
    } else if (i < PREP_X + PREP_QW) {
        int t = i - PREP_X;
        int c = t & (Cc - 1);
        int j = t >> 7;
        qkvwT[(size_t)j * Cc + c] = (_Float16)qkv_w[(size_t)c * JQKV + j];
    } else if (i < PREP_TOT) {
        int t = i - PREP_X - PREP_QW;
        int c = t & (Cc - 1);
        int j = t >> 7;
        gatewT[(size_t)j * Cc + c] = (_Float16)gate_w[(size_t)c * JOUT + j];
    }
}

// ---------------------------------------------------------------------------
// Kernel 2: qkv = xt @ qkv_w + qkv_b, scatter into Q, K (row-major [bh][n][d],
// Q pre-scaled by d^-0.5) and Vt (transposed [bh][d][n]).
// One wave per 16x16 tile: 512 n-tiles x 24 j-tiles = 12288 waves.
// ---------------------------------------------------------------------------
__global__ __launch_bounds__(256) void qkv_kernel(const _Float16* __restrict__ xt,
                                                  const _Float16* __restrict__ qkvwT,
                                                  const float* __restrict__ qkv_b,
                                                  _Float16* __restrict__ Q,
                                                  _Float16* __restrict__ K,
                                                  _Float16* __restrict__ Vt) {
    const int wave = (blockIdx.x * 256 + threadIdx.x) >> 5;
    const int lane = threadIdx.x & 31;
    const int jt = wave % 24;
    const int bn = wave / 24;          // b*64 + ntile
    const int b  = bn >> 6;
    const int n0 = (bn & 63) * 16;
    const int j0 = jt * 16;

    const _Float16* arow = xt + (size_t)b * Nn * Cc;
    f8v acc = fzero8();
#pragma unroll
    for (int kk = 0; kk < Cc; kk += 32) {
        h16 a = load_a_frag(arow, Cc, n0, kk, lane);
        h16 w = load_bT_frag(qkvwT, Cc, j0, kk, lane);
        acc = wmma_f16(a, w, acc);
    }

    const int l16 = lane & 15, g = lane >> 4;
    const int j = j0 + l16;
    const float bias = qkv_b[j];
    const int h = j / 96, t = j % 96, kind = t / 32, dd = t % 32;
    const int bh = b * HEADS + h;

    if (kind == 0) {
        const float qs = 0.17677669529663689f;   // 32^-0.5 folded into Q
#pragma unroll
        for (int r = 0; r < 8; ++r) {
            int n = n0 + 8 * g + r;
            Q[((size_t)bh * Nn + n) * Dd + dd] = (_Float16)((acc[r] + bias) * qs);
        }
    } else if (kind == 1) {
#pragma unroll
        for (int r = 0; r < 8; ++r) {
            int n = n0 + 8 * g + r;
            K[((size_t)bh * Nn + n) * Dd + dd] = (_Float16)(acc[r] + bias);
        }
    } else {
        h8 v;
#pragma unroll
        for (int r = 0; r < 8; ++r) v[r] = (_Float16)(acc[r] + bias);
        *(h8*)(Vt + ((size_t)bh * Dd + dd) * Nn + n0 + 8 * g) = v;  // contiguous 16B
    }
}

// ---------------------------------------------------------------------------
// Kernel 3: flash attention. One wave per (b,h,16-row q tile): 2048 waves.
// Per 32-key chunk: 2 WMMA (QK^T) + online softmax + P through LDS + 2 WMMA (PV).
// ---------------------------------------------------------------------------
__global__ __launch_bounds__(256) void attn_kernel(const _Float16* __restrict__ Q,
                                                   const _Float16* __restrict__ K,
                                                   const _Float16* __restrict__ Vt,
                                                   _Float16* __restrict__ feats) {
    __shared__ __align__(16) _Float16 P[8][16][32];   // 1KB per wave

    const int wid  = threadIdx.x >> 5;
    const int lane = threadIdx.x & 31;
    const int wave = blockIdx.x * 8 + wid;
    const int bh = wave >> 6;               // 0..31
    const int q0 = (wave & 63) * 16;
    const int b = bh >> 2, h = bh & 3;
    const int l16 = lane & 15, g = lane >> 4;

    const _Float16* qb = Q  + (size_t)bh * Nn * Dd;
    const _Float16* kb = K  + (size_t)bh * Nn * Dd;
    const _Float16* vb = Vt + (size_t)bh * Dd * Nn;

    const h16 aq = load_a_frag(qb, Dd, q0, 0, lane);  // whole Q row block (d=32)

    f8v o0 = fzero8(), o1 = fzero8();
    float m[8], l[8];
#pragma unroll
    for (int r = 0; r < 8; ++r) { m[r] = -1e30f; l[r] = 0.0f; }

    for (int kk = 0; kk < Nn; kk += 32) {
        h16 kb0 = load_bT_frag(kb, Dd, kk,      0, lane);
        h16 kb1 = load_bT_frag(kb, Dd, kk + 16, 0, lane);
        f8v s0 = wmma_f16(aq, kb0, fzero8());   // logits, keys kk..kk+15
        f8v s1 = wmma_f16(aq, kb1, fzero8());   // logits, keys kk+16..kk+31

#pragma unroll
        for (int r = 0; r < 8; ++r) {
            float mx = fmaxf(s0[r], s1[r]);
#pragma unroll
            for (int msk = 1; msk < 16; msk <<= 1)
                mx = fmaxf(mx, __shfl_xor(mx, msk, 16));
            float mn = fmaxf(m[r], mx);
            float alpha = __expf(m[r] - mn);
            float p0 = __expf(s0[r] - mn);
            float p1 = __expf(s1[r] - mn);
            float rs = p0 + p1;
#pragma unroll
            for (int msk = 1; msk < 16; msk <<= 1)
                rs += __shfl_xor(rs, msk, 16);
            l[r] = l[r] * alpha + rs;
            m[r] = mn;
            o0[r] *= alpha;
            o1[r] *= alpha;
            P[wid][8 * g + r][l16]      = (_Float16)p0;   // C-layout -> LDS
            P[wid][8 * g + r][16 + l16] = (_Float16)p1;
        }
        // LDS is in-order per wave; wait stores + compiler barrier, then re-read
        // in A-fragment layout.
        asm volatile("s_wait_dscnt 0x0" ::: "memory");
        h16 pa = load_a_frag(&P[wid][0][0], 32, 0, 0, lane);
        asm volatile("" ::: "memory");

        h16 v0 = load_bT_frag(vb, Nn, 0,  kk, lane);  // out cols d=0..15
        h16 v1 = load_bT_frag(vb, Nn, 16, kk, lane);  // out cols d=16..31
        o0 = wmma_f16(pa, v0, o0);
        o1 = wmma_f16(pa, v1, o1);
    }

    _Float16* fb = feats + ((size_t)b * Nn + q0) * CH + h * Dd;
#pragma unroll
    for (int r = 0; r < 8; ++r) {
        float inv = 1.0f / l[r];
        int n = 8 * g + r;
        fb[(size_t)n * CH + l16]      = (_Float16)(o0[r] * inv);
        fb[(size_t)n * CH + 16 + l16] = (_Float16)(o1[r] * inv);
    }
}

// ---------------------------------------------------------------------------
// Kernel 4: out = feats @ gate_w + gate_b; y = x + val * sigmoid(gate),
// written as [B, C, H, W] f32. One wave computes a 16n x 16c tile of BOTH the
// val (j=c) and gate (j=c+128) columns: 8 WMMA per wave, 4096 waves.
// ---------------------------------------------------------------------------
__global__ __launch_bounds__(256) void out_kernel(const _Float16* __restrict__ feats,
                                                  const _Float16* __restrict__ gatewT,
                                                  const float* __restrict__ gate_b,
                                                  const float* __restrict__ x,
                                                  float* __restrict__ out) {
    const int wave = (blockIdx.x * 256 + threadIdx.x) >> 5;
    const int lane = threadIdx.x & 31;
    const int ct = wave & 7;
    const int bn = wave >> 3;           // b*64 + ntile
    const int b  = bn >> 6;
    const int n0 = (bn & 63) * 16;
    const int c0 = ct * 16;

    const _Float16* fb = feats + (size_t)b * Nn * CH;
    f8v av = fzero8(), ag = fzero8();
#pragma unroll
    for (int kk = 0; kk < CH; kk += 32) {
        h16 a  = load_a_frag(fb, CH, n0, kk, lane);
        h16 bv = load_bT_frag(gatewT, CH, c0,       kk, lane);
        h16 bg = load_bT_frag(gatewT, CH, 128 + c0, kk, lane);
        av = wmma_f16(a, bv, av);
        ag = wmma_f16(a, bg, ag);
    }

    const int l16 = lane & 15, g = lane >> 4;
    const int c = c0 + l16;
    const float biasv = gate_b[c];
    const float biasg = gate_b[128 + c];

    const size_t base = ((size_t)b * Cc + c) * Nn + n0 + 8 * g;  // rows n contiguous
    f4v x0 = *(const f4v*)(x + base);
    f4v x1 = *(const f4v*)(x + base + 4);
    f4v y0, y1;
#pragma unroll
    for (int r = 0; r < 4; ++r) {
        float gt = ag[r] + biasg;
        y0[r] = x0[r] + (av[r] + biasv) * (1.0f / (1.0f + __expf(-gt)));
    }
#pragma unroll
    for (int r = 4; r < 8; ++r) {
        float gt = ag[r] + biasg;
        y1[r - 4] = x1[r - 4] + (av[r] + biasv) * (1.0f / (1.0f + __expf(-gt)));
    }
    *(f4v*)(out + base)     = y0;
    *(f4v*)(out + base + 4) = y1;
}

// ---------------------------------------------------------------------------
// Host launcher
// ---------------------------------------------------------------------------
extern "C" void kernel_launch(void* const* d_in, const int* in_sizes, int n_in,
                              void* d_out, int out_size, void* d_ws, size_t ws_size,
                              hipStream_t stream) {
    const float* x      = (const float*)d_in[0];
    const float* qkv_w  = (const float*)d_in[1];
    const float* qkv_b  = (const float*)d_in[2];
    const float* gate_w = (const float*)d_in[3];
    const float* gate_b = (const float*)d_in[4];
    float* out = (float*)d_out;

    char* ws = (char*)d_ws;
    _Float16* xt     = (_Float16*)ws;  ws += (size_t)Bb * Nn * Cc * 2;          // 2 MB
    _Float16* qkvwT  = (_Float16*)ws;  ws += (size_t)JQKV * Cc * 2;             // 96 KB
    _Float16* gatewT = (_Float16*)ws;  ws += (size_t)JOUT * Cc * 2;             // 64 KB
    _Float16* Q      = (_Float16*)ws;  ws += (size_t)Bb * HEADS * Nn * Dd * 2;  // 2 MB
    _Float16* K      = (_Float16*)ws;  ws += (size_t)Bb * HEADS * Nn * Dd * 2;  // 2 MB
    _Float16* Vt     = (_Float16*)ws;  ws += (size_t)Bb * HEADS * Dd * Nn * 2;  // 2 MB
    _Float16* feats  = (_Float16*)ws;  ws += (size_t)Bb * Nn * CH * 2;          // 2 MB

    prep_kernel<<<PREP_TOT / 256, 256, 0, stream>>>(x, qkv_w, gate_w, xt, qkvwT, gatewT);
    qkv_kernel<<<(512 * 24) / 8, 256, 0, stream>>>(xt, qkvwT, qkv_b, Q, K, Vt);
    attn_kernel<<<2048 / 8, 256, 0, stream>>>(Q, K, Vt, feats);
    out_kernel<<<4096 / 8, 256, 0, stream>>>(feats, gatewT, gate_b, x, out);
}
